// MambaBlockSVD_4346506904254
// MI455X (gfx1250) — matrix operand order repaired and checked by reading, not compiled, not run on hardware
//
#include <hip/hip_runtime.h>

// ---------------- problem constants ----------------
#define D_MODEL 768
#define DT_RANK 48
#define D_CONV  4
#define D_INNER 1536
#define D_STATE 16
#define R_IN    384
#define R_X     40
#define R_DT    24
#define R_OUT   384
#define BATCH   2
#define SEQ     2048
#define M_TOTAL (BATCH * SEQ)   // 4096 rows through every GEMM
#define LDX     128             // x_dbl leading dim (padded 80 -> 128)

typedef __attribute__((ext_vector_type(16))) _Float16 v16h;
typedef __attribute__((ext_vector_type(8)))  _Float16 v8h;
typedef __attribute__((ext_vector_type(8)))  float    v8f;
typedef unsigned int u32x4 __attribute__((ext_vector_type(4)));
typedef int          i32x8 __attribute__((ext_vector_type(8)));
typedef int          i32x4 __attribute__((ext_vector_type(4)));

// ---------------------------------------------------------------------------
// Pack an f32 weight W[N,K] (optionally row-scaled by S, folding the SVD diag
// into Vh) into an f16 buffer [Npad,Kpad], zero-padded. Npad is always a
// multiple of 64 so the GEMM needs no N guard and TDM tile loads never go OOB.
// ---------------------------------------------------------------------------
__global__ void pack_weight_f16(const float* __restrict__ W,
                                const float* __restrict__ S,
                                _Float16* __restrict__ out,
                                int N, int K, int Npad, int Kpad) {
  int idx = blockIdx.x * blockDim.x + threadIdx.x;
  int total = Npad * Kpad;
  if (idx >= total) return;
  int r = idx / Kpad;
  int c = idx - r * Kpad;
  float v = 0.f;
  if (r < N && c < K) {
    v = W[(size_t)r * K + c];
    if (S) v *= S[r];
  }
  out[idx] = (_Float16)v;
}

__global__ void cast_f32_f16(const float* __restrict__ in,
                             _Float16* __restrict__ out, int n) {
  int i = blockIdx.x * blockDim.x + threadIdx.x;
  if (i < n) out[i] = (_Float16)in[i];
}

// ---------------------------------------------------------------------------
// TDM issue helper: one 2D tile slice [16 rows x 32 halves] of the W panel,
// global (row-major, stride ldw halves) -> LDS, via the Tensor Data Mover.
// D# layout per CDNA5 ISA ch.8 (group0: count/lds/global/type, group1: dims).
// ---------------------------------------------------------------------------
__device__ __forceinline__ void tdm_load_tile16x32(unsigned ldsAddr,
                                                   unsigned long long gAddr,
                                                   int ldw) {
  u32x4 g0;
  g0[0] = 1u;                                   // count=1, user descriptor
  g0[1] = ldsAddr;                              // LDS byte address
  g0[2] = (unsigned)gAddr;                      // global_addr[31:0]
  g0[3] = (unsigned)(gAddr >> 32) | 0x80000000u; // addr[56:32] | type=2
  i32x8 g1;
  g1[0] = 0x10000;                              // data_size=1 (2 bytes)
  g1[1] = (int)(32u << 16);                     // tensor_dim0 = 32 (lo16 @ [31:16])
  g1[2] = (int)(16u << 16);                     // dim0 hi16=0 | tensor_dim1 = 16
  g1[3] = (int)(32u << 16);                     // dim1 hi16=0 | tile_dim0 = 32
  g1[4] = 16;                                   // tile_dim1 = 16, tile_dim2 = 0
  g1[5] = ldw;                                  // tensor_dim0_stride lo32 (elements)
  g1[6] = 0;                                    // stride0 hi16 | stride1 lo16
  g1[7] = 0;                                    // stride1 hi32
  i32x4 z4 = {};
#if __clang_major__ >= 23
  i32x8 z8 = {};
  __builtin_amdgcn_tensor_load_to_lds(g0, g1, z4, z4, z8, 0);
#else
  __builtin_amdgcn_tensor_load_to_lds(g0, g1, z4, z4, 0);
#endif
}

// ---------------------------------------------------------------------------
// WMMA GEMM:  O[M, Npad] = A[M, >=K] (f16, row-major) @ W[Npad, Kpad]^T (f16)
// Block = 128 threads = 4 waves. Block tile 128(M) x 64(N), K-step 32.
// W tile (64x32 f16 = 4KB) streamed into double-buffered LDS by the Tensor
// Data Mover (one 16-row D# per wave, TENSORcnt-tracked), overlapping WMMAs.
// Each wave computes 32x64 (2 A-frags x 4 B-frags = 8 WMMAs per K-step).
// Requires: Npad % 64 == 0, K % 32 == 0, M % 128 == 0, lda/ldw % 8 == 0.
// ---------------------------------------------------------------------------
template <bool STORE_F32>
__global__ void __launch_bounds__(128)
gemm_xwT_wmma(const _Float16* __restrict__ A, int lda,
              const _Float16* __restrict__ W, int ldw,
              _Float16* __restrict__ Oh, float* __restrict__ Of,
              int ldo, int K) {
  __shared__ __align__(16) _Float16 sW[2][64 * 32];   // 2 x 4KB W tiles

  const int tid  = threadIdx.x;
  const int lane = tid & 31;
  const int wave = tid >> 5;
  const int m0   = blockIdx.y * 128 + wave * 32;   // wave's 32 output rows
  const int n0   = blockIdx.x * 64;

  // --- TDM W-tile staging: each wave DMA's rows [wave*16, wave*16+16) ---
  const unsigned long long gbase =
      (unsigned long long)(size_t)(W + (size_t)(n0 + wave * 16) * ldw);
  const unsigned ldsB0 = (unsigned)(size_t)(&sW[0][wave * 16 * 32]);
  const unsigned ldsB1 = (unsigned)(size_t)(&sW[1][wave * 16 * 32]);

  auto issueTile = [&](int buf, int k0) {
    tdm_load_tile16x32(buf ? ldsB1 : ldsB0,
                       gbase + (unsigned long long)(k0 * 2), ldw);
  };

  // A fragment addressing (contiguous k order, same as B):
  //   lanes 0-15: row=lane, halves K 0-15; lanes 16-31: K 16-31
  const int arow = m0 + (lane & 15);
  const int kbA  = (lane >> 4) * 16;
  // B fragment addressing (contiguous k order, same as A): col=lane&15; lanes>=16 hold K 16-31
  const int colB = lane & 15;
  const int kbB  = (lane >> 4) * 16;

  v8f acc[2][4];
#pragma unroll
  for (int mt = 0; mt < 2; ++mt)
#pragma unroll
    for (int nt = 0; nt < 4; ++nt) acc[mt][nt] = {};

  issueTile(0, 0);   // prologue: tile 0 in flight (TENSORcnt=1 per wave)

  for (int k0 = 0, it = 0; k0 < K; k0 += 32, ++it) {
    const int buf = it & 1;
    __syncthreads();                       // everyone done reading buf^1 (iter it-1)
    if (k0 + 32 < K) {
      issueTile(buf ^ 1, k0 + 32);         // prefetch next tile via TDM
      __builtin_amdgcn_s_wait_tensorcnt(1); // my tile-it slice landed; it+1 in flight
    } else {
      __builtin_amdgcn_s_wait_tensorcnt(0);
    }
    __syncthreads();                       // all waves' tile-it slices visible

    const _Float16* ap0 = A + (size_t)arow * lda + k0 + kbA;
    const _Float16* ap1 = ap0 + (size_t)16 * lda;
    v8h a0lo = *(const v8h*)(ap0);
    v8h a0hi = *(const v8h*)(ap0 + 8);
    v8h a1lo = *(const v8h*)(ap1);
    v8h a1hi = *(const v8h*)(ap1 + 8);
    v16h a0 = __builtin_shufflevector(a0lo, a0hi, 0,1,2,3,4,5,6,7,8,9,10,11,12,13,14,15);
    v16h a1 = __builtin_shufflevector(a1lo, a1hi, 0,1,2,3,4,5,6,7,8,9,10,11,12,13,14,15);

    const _Float16* sb = &sW[buf][0];
    v16h b0 = *(const v16h*)(sb + (0 * 16 + colB) * 32 + kbB);
    v16h b1 = *(const v16h*)(sb + (1 * 16 + colB) * 32 + kbB);
    v16h b2 = *(const v16h*)(sb + (2 * 16 + colB) * 32 + kbB);
    v16h b3 = *(const v16h*)(sb + (3 * 16 + colB) * 32 + kbB);

    acc[0][0] = __builtin_amdgcn_wmma_f32_16x16x32_f16(false, a0, false, b0, (short)0, acc[0][0], false, false);
    acc[1][0] = __builtin_amdgcn_wmma_f32_16x16x32_f16(false, a1, false, b0, (short)0, acc[1][0], false, false);
    acc[0][1] = __builtin_amdgcn_wmma_f32_16x16x32_f16(false, a0, false, b1, (short)0, acc[0][1], false, false);
    acc[1][1] = __builtin_amdgcn_wmma_f32_16x16x32_f16(false, a1, false, b1, (short)0, acc[1][1], false, false);
    acc[0][2] = __builtin_amdgcn_wmma_f32_16x16x32_f16(false, a0, false, b2, (short)0, acc[0][2], false, false);
    acc[1][2] = __builtin_amdgcn_wmma_f32_16x16x32_f16(false, a1, false, b2, (short)0, acc[1][2], false, false);
    acc[0][3] = __builtin_amdgcn_wmma_f32_16x16x32_f16(false, a0, false, b3, (short)0, acc[0][3], false, false);
    acc[1][3] = __builtin_amdgcn_wmma_f32_16x16x32_f16(false, a1, false, b3, (short)0, acc[1][3], false, false);
  }

  // Epilogue. C/D layout: lanes 0-15 -> rows +0..7, lanes 16-31 -> rows +8..15,
  // column = lane&15 within each 16-wide N tile. Npad%64==0 -> no guards.
  const int rbase = (lane >> 4) * 8;
  const int cc    = lane & 15;
#pragma unroll
  for (int mt = 0; mt < 2; ++mt) {
#pragma unroll
    for (int nt = 0; nt < 4; ++nt) {
      int n = n0 + nt * 16 + cc;
#pragma unroll
      for (int r = 0; r < 8; ++r) {
        size_t o = (size_t)(m0 + mt * 16 + rbase + r) * ldo + n;
        if (STORE_F32) Of[o] = acc[mt][nt][r];
        else           Oh[o] = (_Float16)acc[mt][nt][r];
      }
    }
  }
}

// ---------------------------------------------------------------------------
// Depthwise causal conv1d (kernel 4) + SiLU over the x half of xz.
// ---------------------------------------------------------------------------
__global__ void conv_silu(const _Float16* __restrict__ xz,
                          const float* __restrict__ cw,
                          const float* __restrict__ cb,
                          _Float16* __restrict__ xconv) {
  int t = blockIdx.x * blockDim.x + threadIdx.x;
  if (t >= M_TOTAL * D_INNER) return;
  int row = t / D_INNER;
  int d   = t - row * D_INNER;
  int b   = row >> 11;        // / SEQ
  int l   = row & (SEQ - 1);
  float acc = cb[d];
#pragma unroll
  for (int k = 0; k < D_CONV; ++k) {
    int ll = l + k - (D_CONV - 1);
    if (ll >= 0)
      acc += (float)xz[(size_t)((b << 11) + ll) * (2 * D_INNER) + d] * cw[d * D_CONV + k];
  }
  float s = acc / (1.f + __expf(-acc));            // SiLU
  xconv[(size_t)row * D_INNER + d] = (_Float16)s;
}

// ---------------------------------------------------------------------------
// Fused selective scan: one thread per (batch, channel). h[16] lives in VGPRs.
// B/C (shared by all 1536 channels per timestep) staged through LDS in chunks
// of 16 timesteps (2 barriers / 16 steps). Fuses softplus(dt)+bias, D*x skip,
// SiLU(z) gating. Output f16 feeds out_proj WMMA GEMM.
// ---------------------------------------------------------------------------
__global__ void __launch_bounds__(256)
ssm_scan_fused(const _Float16* __restrict__ xz,
               const _Float16* __restrict__ xconv,
               const _Float16* __restrict__ xdbl,   // [M,LDX]: dt_r | B | C | 0
               const _Float16* __restrict__ dtlin,  // [M,1536]
               const float* __restrict__ dtb,
               const float* __restrict__ A_log,
               const float* __restrict__ Dp,
               _Float16* __restrict__ yg) {
  int d = blockIdx.x * blockDim.x + threadIdx.x;   // 0..1535
  int b = blockIdx.y;

  float Arow[D_STATE], h[D_STATE];
#pragma unroll
  for (int n = 0; n < D_STATE; ++n) {
    Arow[n] = -__expf(A_log[d * D_STATE + n]);
    h[n] = 0.f;
  }
  const float bias = dtb[d];
  const float dval = Dp[d];

  __shared__ float sBC[16][32];   // 16 timesteps of {B[16] | C[16]}

  for (int l0 = 0; l0 < SEQ; l0 += 16) {
    __syncthreads();
    for (int j = threadIdx.x; j < 512; j += 256) {
      int ll = j >> 5, c = j & 31;
      sBC[ll][c] = (float)xdbl[(size_t)(b * SEQ + l0 + ll) * LDX + DT_RANK + c];
    }
    __syncthreads();

#pragma unroll 4
    for (int t = 0; t < 16; ++t) {
      const size_t row = (size_t)(b * SEQ + l0 + t);

      float dtv = (float)dtlin[row * D_INNER + d] + bias;
      dtv = (dtv > 20.f) ? dtv : log1pf(__expf(dtv));   // softplus
      float xc  = (float)xconv[row * D_INNER + d];
      float dbx = dtv * xc;

      float y = 0.f;
#pragma unroll
      for (int n = 0; n < D_STATE; ++n) {
        h[n] = h[n] * __expf(dtv * Arow[n]) + dbx * sBC[t][n];
        y += h[n] * sBC[t][16 + n];
      }
      y += dval * xc;

      float z = (float)xz[row * (2 * D_INNER) + D_INNER + d];
      float g = z / (1.f + __expf(-z));                 // SiLU(z)
      yg[row * D_INNER + d] = (_Float16)(y * g);
    }
  }
}

// ---------------------------------------------------------------------------
// Host-side orchestration.
// ---------------------------------------------------------------------------
extern "C" void kernel_launch(void* const* d_in, const int* in_sizes, int n_in,
                              void* d_out, int out_size, void* d_ws, size_t ws_size,
                              hipStream_t stream) {
  (void)in_sizes; (void)n_in; (void)out_size; (void)ws_size;

  const float* u      = (const float*)d_in[0];
  const float* inU    = (const float*)d_in[1];
  const float* inS    = (const float*)d_in[2];
  const float* inVh   = (const float*)d_in[3];
  const float* xU     = (const float*)d_in[4];
  const float* xS     = (const float*)d_in[5];
  const float* xVh    = (const float*)d_in[6];
  const float* dtU    = (const float*)d_in[7];
  const float* dtS    = (const float*)d_in[8];
  const float* dtVh   = (const float*)d_in[9];
  const float* dtBias = (const float*)d_in[10];
  const float* outU   = (const float*)d_in[11];
  const float* outS   = (const float*)d_in[12];
  const float* outVh  = (const float*)d_in[13];
  const float* convW  = (const float*)d_in[14];
  const float* convB  = (const float*)d_in[15];
  const float* A_log  = (const float*)d_in[16];
  const float* D_prm  = (const float*)d_in[17];
  float* out = (float*)d_out;

  char* ws = (char*)d_ws;
  size_t off = 0;
  auto alloc = [&](size_t bytes) -> void* {
    void* p = ws + off;
    off += (bytes + 255) & ~(size_t)255;
    return p;
  };
  auto ah = [&](size_t halves) -> _Float16* { return (_Float16*)alloc(halves * 2); };

  // f16 weights (S folded into Vh; Npad rounded up to 64 multiples)
  _Float16* wVh_in  = ah((size_t)R_IN * D_MODEL);          // [384,768]
  _Float16* wU_in   = ah((size_t)(2 * D_INNER) * R_IN);    // [3072,384]
  _Float16* wVh_x   = ah((size_t)64 * D_INNER);            // [64(<-40),1536]
  _Float16* wU_x    = ah((size_t)LDX * 64);                // [128(<-80),64(<-40)]
  _Float16* wVh_dt  = ah((size_t)64 * 64);                 // [64(<-24),64(<-48)]
  _Float16* wU_dt   = ah((size_t)D_INNER * 32);            // [1536,32(<-24)]
  _Float16* wVh_out = ah((size_t)R_OUT * D_INNER);         // [384,1536]
  _Float16* wU_out  = ah((size_t)D_MODEL * R_OUT);         // [768,384]

  // activations
  _Float16* u16   = ah((size_t)M_TOTAL * D_MODEL);
  _Float16* t1    = ah((size_t)M_TOTAL * R_IN);
  _Float16* xzbuf = ah((size_t)M_TOTAL * 2 * D_INNER);
  _Float16* xconv = ah((size_t)M_TOTAL * D_INNER);
  _Float16* t_x   = ah((size_t)M_TOTAL * 64);
  _Float16* xdbl  = ah((size_t)M_TOTAL * LDX);
  _Float16* t_dt  = ah((size_t)M_TOTAL * 64);
  _Float16* dtlin = ah((size_t)M_TOTAL * D_INNER);
  _Float16* ygate = ah((size_t)M_TOTAL * D_INNER);
  _Float16* t_o   = ah((size_t)M_TOTAL * R_OUT);

  auto pack = [&](const float* W, const float* S, _Float16* o,
                  int N, int K, int Np, int Kp) {
    int tot = Np * Kp;
    pack_weight_f16<<<(tot + 255) / 256, 256, 0, stream>>>(W, S, o, N, K, Np, Kp);
  };
  auto gemm = [&](const _Float16* A, int lda, const _Float16* W, int ldw, int Npad,
                  _Float16* Oh, float* Of, int ldo, int K) {
    dim3 g(Npad / 64, M_TOTAL / 128);
    if (Of) gemm_xwT_wmma<true ><<<g, 128, 0, stream>>>(A, lda, W, ldw, nullptr, Of, ldo, K);
    else    gemm_xwT_wmma<false><<<g, 128, 0, stream>>>(A, lda, W, ldw, Oh, nullptr, ldo, K);
  };

  // 0) weight packing (S folded into Vh rows)
  pack(inVh,  inS,  wVh_in,  R_IN,        D_MODEL, R_IN,        D_MODEL);
  pack(inU,   0,    wU_in,   2 * D_INNER, R_IN,    2 * D_INNER, R_IN);
  pack(xVh,   xS,   wVh_x,   R_X,         D_INNER, 64,          D_INNER);
  pack(xU,    0,    wU_x,    DT_RANK + 2 * D_STATE, R_X, LDX,   64);
  pack(dtVh,  dtS,  wVh_dt,  R_DT,        DT_RANK, 64,          64);
  pack(dtU,   0,    wU_dt,   D_INNER,     R_DT,    D_INNER,     32);
  pack(outVh, outS, wVh_out, R_OUT,       D_INNER, R_OUT,       D_INNER);
  pack(outU,  0,    wU_out,  D_MODEL,     R_OUT,   D_MODEL,     R_OUT);

  // 1) u -> f16
  {
    int n = M_TOTAL * D_MODEL;
    cast_f32_f16<<<(n + 255) / 256, 256, 0, stream>>>(u, u16, n);
  }

  // 2) in_proj: t1 = u @ (S*Vh)^T ; xz = t1 @ U^T
  gemm(u16, D_MODEL, wVh_in, D_MODEL, R_IN, t1, nullptr, R_IN, D_MODEL);
  gemm(t1, R_IN, wU_in, R_IN, 2 * D_INNER, xzbuf, nullptr, 2 * D_INNER, R_IN);

  // 3) depthwise causal conv + SiLU on x half
  {
    int n = M_TOTAL * D_INNER;
    conv_silu<<<(n + 255) / 256, 256, 0, stream>>>(xzbuf, convW, convB, xconv);
  }

  // 4) x_proj: t_x = xconv @ (S*Vh)^T (N 40->64); xdbl = t_x @ U^T (N 80->128, K 40->64)
  gemm(xconv, D_INNER, wVh_x, D_INNER, 64, t_x, nullptr, 64, D_INNER);
  gemm(t_x, 64, wU_x, 64, LDX, xdbl, nullptr, LDX, 64);

  // 5) dt_proj on dt_r = xdbl[:, :48] (zero weight cols 48..63 mask out B/C):
  gemm(xdbl, LDX, wVh_dt, 64, 64, t_dt, nullptr, 64, 64);
  gemm(t_dt, 64, wU_dt, 32, D_INNER, dtlin, nullptr, D_INNER, 32);

  // 6) fused selective scan + softplus + D-skip + SiLU(z) gating
  {
    dim3 g(D_INNER / 256, BATCH);
    ssm_scan_fused<<<g, 256, 0, stream>>>(xzbuf, xconv, xdbl, dtlin,
                                          dtBias, A_log, D_prm, ygate);
  }

  // 7) out_proj: t_o = y @ (S*Vh)^T ; out(f32) = t_o @ U^T
  gemm(ygate, D_INNER, wVh_out, D_INNER, R_OUT, t_o, nullptr, R_OUT, D_INNER);
  gemm(t_o, R_OUT, wU_out, R_OUT, D_MODEL, nullptr, out, D_MODEL, R_OUT);
}
